// GNNModel_28458453303300
// MI455X (gfx1250) — matrix operand order repaired and checked by reading, not compiled, not run on hardware
//
#include <hip/hip_runtime.h>
#include <hip/hip_bf16.h>

// ---------------- problem constants ----------------
constexpr int EMB   = 256;
constexpr int Bc    = 256;
constexpr int Ec    = 100000;
constexpr int TOPK  = 64;
constexpr int NPREV = 50000;
constexpr int NENT  = 20000;
constexpr int NTIME = 2000;
constexpr int Mk    = Bc * TOPK;   // 16384 selected rows
constexpr int RELP  = 208;         // 201 rel rows padded to multiple of 16
constexpr int BPAD  = 8;           // LDS row pad (halves): 4-bank lane skew

// ---------------- WMMA types ----------------
typedef __attribute__((ext_vector_type(16))) __bf16 v16bf;
typedef __attribute__((ext_vector_type(8)))  float  v8f;

struct BFPack { unsigned int w[8]; };  // 32B = 16 bf16 per lane

__device__ inline unsigned short f2bf(float f) {
    unsigned int u = __float_as_uint(f);
    unsigned int r = (u + 0x7FFFu + ((u >> 16) & 1u)) >> 16;
    return (unsigned short)r;
}
__device__ inline float sigm(float x) { return 1.0f / (1.0f + __expf(-x)); }
__device__ inline float wred(float v) {
#pragma unroll
    for (int o = 16; o > 0; o >>= 1) v += __shfl_xor(v, o, 32);
    return v;
}

// A-matrix 16x32 bf16 tile loader from global (lane = M row; K contiguous: k = 16*(lane/16) + s)
__device__ inline v16bf load_a16(const unsigned short* rowp, int k0, int lane) {
    int base = k0 + ((lane & 16) ? 16 : 0);
    const uint4* a = (const uint4*)(rowp + base);
    const uint4* b = (const uint4*)(rowp + base + 8);
    uint4 lo = *a, hi = *b;
    BFPack t;
    t.w[0] = lo.x; t.w[1] = lo.y; t.w[2] = lo.z; t.w[3] = lo.w;
    t.w[4] = hi.x; t.w[5] = hi.y; t.w[6] = hi.z; t.w[7] = hi.w;
    return __builtin_bit_cast(v16bf, t);
}
// B-matrix 32x16 bf16 tile loader from LDS (lane = N col; 2x ds_load_b128)
__device__ inline v16bf load_b16_lds(const unsigned short* rowp, int koff) {
    const uint4* a = (const uint4*)(rowp + koff);
    uint4 lo = a[0], hi = a[1];
    BFPack t;
    t.w[0] = lo.x; t.w[1] = lo.y; t.w[2] = lo.z; t.w[3] = lo.w;
    t.w[4] = hi.x; t.w[5] = hi.y; t.w[6] = hi.z; t.w[7] = hi.w;
    return __builtin_bit_cast(v16bf, t);
}

// ===== GEMM: Y[M,N] = X[M,K] * W[N,K]^T (+bias), bf16 in, f32 out ==========
// Block = 8 waves sharing one 64-column B panel, staged into dynamic LDS via
// GLOBAL_LOAD_ASYNC_TO_LDS_B128 (ASYNCcnt). Rows padded to Kdim+8 halves so
// the 16 lanes of each half-wave hit disjoint 4-bank groups (conflict-free
// ds_load_b128). Each wave computes a 16x64 tile (4 accumulators) reusing
// every A fragment 4x.
__global__ __launch_bounds__(256) void gemm_bf16_wmma(
    const unsigned short* __restrict__ X, const unsigned short* __restrict__ W,
    const float* __restrict__ bias, float* __restrict__ Y,
    int Mdim, int Ndim, int Kdim)
{
    extern __shared__ unsigned short sB[];     // 64 * (Kdim+BPAD) halves
    int pitch  = Kdim + BPAD;
    int tid    = threadIdx.x;
    int lane   = tid & 31;
    int waveId = tid >> 5;
    int nBase  = blockIdx.y * 64;

    // ---- async-stage the B panel: 64 x Kdim bf16 -> LDS (padded pitch) ----
    int chunksPerRow = Kdim >> 3;              // 16B chunks per row
    int totalChunks  = 64 * chunksPerRow;      // multiple of 256
    for (int c = tid; c < totalChunks; c += 256) {
        int rw = c / chunksPerRow;
        int ck = (c % chunksPerRow) * 8;
        const unsigned short* g = W + (size_t)(nBase + rw) * Kdim + ck;
        unsigned lds_off = (unsigned)(size_t)&sB[rw * pitch + ck];
        unsigned long long ga = (unsigned long long)(size_t)g;
        asm volatile("global_load_async_to_lds_b128 %0, %1, off"
                     :: "v"(lds_off), "v"(ga) : "memory");
    }
    asm volatile("s_wait_asynccnt 0x0" ::: "memory");
    __syncthreads();

    // ---- each wave: one 16-row M tile x this block's 64-col panel ----
    int mTiles = (Mdim + 15) >> 4;
    int mt = blockIdx.x * 8 + waveId;
    if (mt >= mTiles) mt = mTiles - 1;         // clamp: duplicate identical stores
    int m0 = mt << 4;
    int rA = m0 + (lane & 15); if (rA >= Mdim) rA = Mdim - 1;
    const unsigned short* xa = X + (size_t)rA * Kdim;
    const unsigned short* r0 = &sB[( 0 + (lane & 15)) * pitch];
    const unsigned short* r1 = &sB[(16 + (lane & 15)) * pitch];
    const unsigned short* r2 = &sB[(32 + (lane & 15)) * pitch];
    const unsigned short* r3 = &sB[(48 + (lane & 15)) * pitch];
    int bsel = (lane & 16) ? 16 : 0;

    v8f c0 = {}, c1 = {}, c2 = {}, c3 = {};
    for (int k0 = 0; k0 < Kdim; k0 += 32) {
        v16bf a  = load_a16(xa, k0, lane);
        v16bf b0 = load_b16_lds(r0, k0 + bsel);
        v16bf b1 = load_b16_lds(r1, k0 + bsel);
        v16bf b2 = load_b16_lds(r2, k0 + bsel);
        v16bf b3 = load_b16_lds(r3, k0 + bsel);
        c0 = __builtin_amdgcn_wmma_f32_16x16x32_bf16(false, a, false, b0, (short)0, c0, false, false);
        c1 = __builtin_amdgcn_wmma_f32_16x16x32_bf16(false, a, false, b1, (short)0, c1, false, false);
        c2 = __builtin_amdgcn_wmma_f32_16x16x32_bf16(false, a, false, b2, (short)0, c2, false, false);
        c3 = __builtin_amdgcn_wmma_f32_16x16x32_bf16(false, a, false, b3, (short)0, c3, false, false);
    }
    int rbase = m0 + ((lane & 16) ? 8 : 0);
#pragma unroll
    for (int s = 0; s < 4; ++s) {
        int col = nBase + s * 16 + (lane & 15);
        float bv = bias ? bias[col] : 0.0f;
        v8f cc = (s == 0) ? c0 : (s == 1) ? c1 : (s == 2) ? c2 : c3;
#pragma unroll
        for (int j = 0; j < 8; ++j) {
            int row = rbase + j;
            if (row < Mdim) Y[(size_t)row * Ndim + col] = cc[j] + bv;
        }
    }
}

// ================= small utility kernels =================
__global__ void zero_f32(float* p, size_t n) {
    size_t i = (size_t)blockIdx.x * blockDim.x + threadIdx.x;
    size_t st = (size_t)gridDim.x * blockDim.x;
    for (; i < n; i += st) p[i] = 0.0f;
}
__global__ void cvt_bf16(const float* __restrict__ s, unsigned short* __restrict__ d, size_t n) {
    size_t i = (size_t)blockIdx.x * blockDim.x + threadIdx.x;
    size_t st = (size_t)gridDim.x * blockDim.x;
    for (; i < n; i += st) d[i] = f2bf(s[i]);
}
__global__ void cvt_split_bf16(const float* __restrict__ s, unsigned short* __restrict__ d,
                               int rows, int cols, int ld, int off) {
    size_t n = (size_t)rows * cols;
    size_t i = (size_t)blockIdx.x * blockDim.x + threadIdx.x;
    size_t st = (size_t)gridDim.x * blockDim.x;
    for (; i < n; i += st) {
        int r = (int)(i / cols), c = (int)(i % cols);
        d[i] = f2bf(s[(size_t)r * ld + off + c]);
    }
}
__global__ void build_relall(const float* __restrict__ zr, const float* __restrict__ re,
                             unsigned short* __restrict__ d) {
    int i = blockIdx.x * blockDim.x + threadIdx.x;
    if (i >= RELP * 256) return;
    int r = i >> 8, c = i & 255;
    float v = (r == 0) ? zr[c] : ((r <= 200) ? re[(size_t)(r - 1) * 256 + c] : 0.0f);
    d[i] = f2bf(v);
}
// out[row] = sum_d a[d]*T[row*256+d] (one wave / row)
__global__ __launch_bounds__(256) void dot_rows(const float* __restrict__ T,
                                                const float* __restrict__ a,
                                                float* __restrict__ out, int rows) {
    int wave = blockIdx.x * 8 + (threadIdx.x >> 5);
    int lane = threadIdx.x & 31;
    if (wave >= rows) return;
    float s = 0.f;
    for (int d = lane; d < 256; d += 32) s += a[d] * T[(size_t)wave * 256 + d];
    s = wred(s);
    if (lane == 0) out[wave] = s;
}
__global__ void consts_kernel(const float* a1w, const float* a1b, const float* wmb,
                              const float* leb, float* consts) {
    int l = threadIdx.x;
    float s = 0.f;
    for (int d = l; d < 256; d += 32) s += a1w[256 + d] * wmb[d] + a1w[512 + d] * leb[d];
    s = wred(s);
    if (l == 0) consts[0] = s + a1b[0];
}

// ================= per-edge attention (GRU + factored att1) =================
__global__ __launch_bounds__(256) void attention_kernel(
    const int* __restrict__ facts, const float* __restrict__ gi_tab,
    const float* __restrict__ gh_tab, const float* __restrict__ hidden_rel,
    const float* __restrict__ att2w, const float* __restrict__ att2b,
    const float* __restrict__ qdot, const float* __restrict__ dmA,
    const float* __restrict__ dmB, const float* __restrict__ dhtA,
    const float* __restrict__ dhtB, const float* __restrict__ consts,
    float* __restrict__ attention, int E_)
{
    int wave = blockIdx.x * 8 + (threadIdx.x >> 5);
    int lane = threadIdx.x & 31;
    if (wave >= E_) return;
    const int* f = facts + (size_t)wave * 6;
    int b = f[0], n = f[1], r1 = f[3] + 1, tt = f[4], ts = f[5];
    const float* gi = gi_tab + (size_t)r1 * 768;
    const float* gh = gh_tab + (size_t)n * 768;
    const float* hp = hidden_rel + (size_t)n * 256;
    float s2 = 0.f;
#pragma unroll
    for (int q = 0; q < 8; ++q) {
        int d = lane + 32 * q;
        float rg = sigm(gi[d] + gh[d]);
        float zg = sigm(gi[256 + d] + gh[256 + d]);
        float ng = tanhf(gi[512 + d] + rg * gh[512 + d]);
        float hnew = (1.f - zg) * ng + zg * hp[d];
        s2 += att2w[d] * hnew;
    }
    s2 = wred(s2);
    if (lane == 0) {
        float a2 = sigm(s2 + att2b[0]);
        float a1 = sigm(qdot[b] + dmA[n] + dmB[r1] + dhtA[tt] + dhtB[ts] + consts[0]);
        attention[wave] = 0.5f * (a1 + a2);
    }
}

// ================= bucketing by b, per-batch top-64 =================
__global__ void count_kernel(const int* __restrict__ facts, int* counts, int E_) {
    int i = blockIdx.x * blockDim.x + threadIdx.x;
    if (i < E_) atomicAdd(&counts[facts[(size_t)i * 6]], 1);
}
__global__ __launch_bounds__(256) void scan256(const int* counts, int* starts) {
    __shared__ int s[256];
    int t = threadIdx.x;
    s[t] = counts[t];
    __syncthreads();
    for (int off = 1; off < 256; off <<= 1) {
        int x = (t >= off) ? s[t - off] : 0;
        __syncthreads();
        s[t] += x;
        __syncthreads();
    }
    starts[t] = s[t] - counts[t];
}
__global__ void scatter_kernel(const int* __restrict__ facts, const int* __restrict__ starts,
                               int* cursor, int* bucket, int E_) {
    int i = blockIdx.x * blockDim.x + threadIdx.x;
    if (i >= E_) return;
    int b = facts[(size_t)i * 6];
    int pos = starts[b] + atomicAdd(&cursor[b], 1);
    bucket[pos] = i;
}
// one block per batch: bitonic sort <=512 (att desc, idx asc), keep top 64
__global__ __launch_bounds__(256) void topk_kernel(
    const int* __restrict__ bucket, const int* __restrict__ starts,
    const int* __restrict__ counts, const float* __restrict__ attention,
    int* __restrict__ sel)
{
    __shared__ float sa[512];
    __shared__ int   si[512];
    int b = blockIdx.x, t = threadIdx.x;
    int cnt = counts[b]; if (cnt > 512) cnt = 512;
    int st = starts[b];
    for (int p = t; p < 512; p += 256) {
        if (p < cnt) { int e = bucket[st + p]; sa[p] = attention[e]; si[p] = e; }
        else         { sa[p] = -3.0e38f; si[p] = 0x7FFFFFFF; }
    }
    __syncthreads();
    for (int kk = 2; kk <= 512; kk <<= 1)
        for (int j = kk >> 1; j > 0; j >>= 1) {
            for (int i = t; i < 512; i += 256) {
                int p = i ^ j;
                if (p > i) {
                    bool up = ((i & kk) == 0);
                    bool less_ip = (sa[i] > sa[p]) || (sa[i] == sa[p] && si[i] < si[p]);
                    bool sw = up ? (!less_ip) : less_ip;
                    if (sw) {
                        float tf = sa[i]; sa[i] = sa[p]; sa[p] = tf;
                        int   ti = si[i]; si[i] = si[p]; si[p] = ti;
                    }
                }
            }
            __syncthreads();
        }
    if (t < TOPK) sel[b * TOPK + t] = si[t];
}

// ================= selected-row expansion (recompute GRU, message, key) =====
__global__ __launch_bounds__(256) void selrow_kernel(
    const int* __restrict__ sel, const int* __restrict__ facts,
    const float* __restrict__ gi_tab, const float* __restrict__ gh_tab,
    const float* __restrict__ hidden_rel, const float* __restrict__ msgA,
    const float* __restrict__ msgB, const float* __restrict__ wmb,
    const float* __restrict__ attention, const float* __restrict__ att_agg,
    float* __restrict__ hrnew, float* __restrict__ msg_sel,
    float* __restrict__ att_sel, int* __restrict__ b_sel,
    unsigned long long* __restrict__ keys)
{
    int wave = blockIdx.x * 8 + (threadIdx.x >> 5);
    int lane = threadIdx.x & 31;
    if (wave >= Mk) return;
    int e = sel[wave];
    const int* f = facts + (size_t)e * 6;
    int b = f[0], n = f[1], r1 = f[3] + 1, tt = f[4], ts = f[5];
    const float* gi = gi_tab + (size_t)r1 * 768;
    const float* gh = gh_tab + (size_t)n * 768;
    const float* hp = hidden_rel + (size_t)n * 256;
#pragma unroll
    for (int q = 0; q < 8; ++q) {
        int d = lane + 32 * q;
        float rg = sigm(gi[d] + gh[d]);
        float zg = sigm(gi[256 + d] + gh[256 + d]);
        float ng = tanhf(gi[512 + d] + rg * gh[512 + d]);
        hrnew[(size_t)wave * 256 + d] = (1.f - zg) * ng + zg * hp[d];
        msg_sel[(size_t)wave * 256 + d] =
            msgA[(size_t)n * 256 + d] + msgB[(size_t)r1 * 256 + d] + wmb[d];
    }
    if (lane == 0) {
        att_sel[wave] = attention[e] * att_agg[n];
        b_sel[wave] = b;
        keys[wave] = ((unsigned long long)b << 32) | ((unsigned long long)tt << 16) |
                     (unsigned long long)ts;
    }
}

__global__ void den_kernel(const int* b_sel, const float* att_sel, float* den) {
    int i = blockIdx.x * blockDim.x + threadIdx.x;
    if (i < Mk) atomicAdd(&den[b_sel[i]], att_sel[i]);
}
__global__ void norm_kernel(const int* b_sel, const float* att_sel, const float* den,
                            float* att_norm) {
    int i = blockIdx.x * blockDim.x + threadIdx.x;
    if (i >= Mk) return;
    float d = den[b_sel[i]];
    att_norm[i] = (d != 0.0f) ? att_sel[i] / d : 0.0f;
}

// ========= single-workgroup 16384-key bitonic sort + unique (196KB LDS) =====
__global__ __launch_bounds__(1024) void sort_unique_kernel(
    const unsigned long long* __restrict__ keys_in, int* __restrict__ tidx,
    int* __restrict__ tailn, float* __restrict__ tailn_f)
{
    extern __shared__ char smem[];
    unsigned long long* k = (unsigned long long*)smem;         // 16384*8 = 128KB
    int* v  = (int*)(smem + (size_t)Mk * 8);                   // 16384*4 =  64KB
    int* ps = (int*)(smem + (size_t)Mk * 8 + (size_t)Mk * 4);  //  1024*4 =   4KB
    int t = threadIdx.x;
    for (int p = t; p < Mk; p += 1024) { k[p] = keys_in[p]; v[p] = p; }
    __syncthreads();
    for (int kk = 2; kk <= Mk; kk <<= 1)
        for (int j = kk >> 1; j > 0; j >>= 1) {
            for (int i = t; i < Mk; i += 1024) {
                int p = i ^ j;
                if (p > i) {
                    bool up = ((i & kk) == 0);
                    bool sw = up ? (k[i] > k[p]) : (k[i] < k[p]);
                    if (sw) {
                        unsigned long long tk = k[i]; k[i] = k[p]; k[p] = tk;
                        int tv = v[i]; v[i] = v[p]; v[p] = tv;
                    }
                }
            }
            __syncthreads();
        }
    // head flags -> block-wide scan -> unique ids
    const int CH = Mk / 1024;   // 16 per thread
    int s = 0;
#pragma unroll
    for (int q = 0; q < CH; ++q) {
        int p = t * CH + q;
        s += (p == 0) || (k[p] != k[p - 1]);
    }
    ps[t] = s;
    __syncthreads();
    for (int off = 1; off < 1024; off <<= 1) {
        int x = (t >= off) ? ps[t - off] : 0;
        __syncthreads();
        ps[t] += x;
        __syncthreads();
    }
    int run = ps[t] - s;   // exclusive base
#pragma unroll
    for (int q = 0; q < CH; ++q) {
        int p = t * CH + q;
        int fl = (p == 0) || (k[p] != k[p - 1]);
        run += fl;
        int uid = run - 1;
        tidx[v[p]] = uid;
        if (fl) {
            unsigned long long key = k[p];
            int bb = (int)(key >> 32);
            int t2 = (int)((key >> 16) & 0xFFFF);
            int t3 = (int)(key & 0xFFFF);
            tailn[uid * 3 + 0] = bb;  tailn[uid * 3 + 1] = t2;  tailn[uid * 3 + 2] = t3;
            tailn_f[uid * 3 + 0] = (float)bb;
            tailn_f[uid * 3 + 1] = (float)t2;
            tailn_f[uid * 3 + 2] = (float)t3;
        }
    }
}

// ================= segment aggregation (float atomics) =================
__global__ __launch_bounds__(256) void aggregate_kernel(
    const int* __restrict__ tidx, const float* __restrict__ att_norm,
    const float* __restrict__ hrnew, const float* __restrict__ msg_sel,
    float* __restrict__ msg_agg, float* __restrict__ hro, float* __restrict__ attagg)
{
    int wave = blockIdx.x * 8 + (threadIdx.x >> 5);
    int lane = threadIdx.x & 31;
    if (wave >= Mk) return;
    int u = tidx[wave];
    float a = att_norm[wave];
#pragma unroll
    for (int q = 0; q < 8; ++q) {
        int d = lane + 32 * q;
        atomicAdd(&msg_agg[(size_t)u * 256 + d], a * msg_sel[(size_t)wave * 256 + d]);
        atomicAdd(&hro[(size_t)u * 256 + d], a * hrnew[(size_t)wave * 256 + d]);
    }
    if (lane == 0) atomicAdd(&attagg[u], a);
}

// build bf16 input [Mk,512] = [msg_agg | tail_emd] for final WMMA GEMM
__global__ void build_xh(const float* __restrict__ msg_agg, const float* __restrict__ htA,
                         const float* __restrict__ htB, const float* __restrict__ leb,
                         const int* __restrict__ tailn, unsigned short* __restrict__ xh)
{
    int i = blockIdx.x * blockDim.x + threadIdx.x;
    if (i >= Mk * 256) return;
    int u = i >> 8, d = i & 255;
    xh[(size_t)u * 512 + d] = f2bf(msg_agg[i]);
    int tu = tailn[u * 3 + 1], tsu = tailn[u * 3 + 2];
    xh[(size_t)u * 512 + 256 + d] =
        f2bf(htA[(size_t)tu * 256 + d] + htB[(size_t)tsu * 256 + d] + leb[d]);
}

// ==========================================================================
extern "C" void kernel_launch(void* const* d_in, const int* in_sizes, int n_in,
                              void* d_out, int out_size, void* d_ws, size_t ws_size,
                              hipStream_t stream)
{
    const float* query_emd = (const float*)d_in[1];
    const int*   facts     = (const int*)d_in[2];
    const float* ent_emd   = (const float*)d_in[3];
    const float* rel_emd   = (const float*)d_in[4];
    const float* hidden_node = (const float*)d_in[5];
    const float* hidden_rel  = (const float*)d_in[6];
    const float* att_agg   = (const float*)d_in[8];
    const float* time_emd  = (const float*)d_in[9];
    const float* zero_rel  = (const float*)d_in[10];
    const float* Wmsg      = (const float*)d_in[11];
    const float* Wmsg_b    = (const float*)d_in[12];
    const float* a1w       = (const float*)d_in[13];
    const float* a1b       = (const float*)d_in[14];
    const float* a2w       = (const float*)d_in[15];
    const float* a2b       = (const float*)d_in[16];
    const float* Whw       = (const float*)d_in[17];
    const float* Whb       = (const float*)d_in[18];
    const float* lets_w    = (const float*)d_in[19];
    const float* lets_b    = (const float*)d_in[20];
    const float* lrts_w    = (const float*)d_in[21];
    const float* lrts_b    = (const float*)d_in[22];
    const float* gwih      = (const float*)d_in[25];
    const float* gwhh      = (const float*)d_in[26];
    const float* gbih      = (const float*)d_in[27];
    const float* gbhh      = (const float*)d_in[28];

    // ---- workspace carve-up (256B aligned) ----
    char* p = (char*)d_ws;
    auto alloc = [&](size_t bytes) -> void* {
        void* r = (void*)p;
        p += (bytes + 255) & ~size_t(255);
        return r;
    };
    unsigned short* hrl_bf   = (unsigned short*)alloc((size_t)NPREV * 256 * 2);
    unsigned short* hnode_bf = (unsigned short*)alloc((size_t)NPREV * 256 * 2);
    unsigned short* ent_bf   = (unsigned short*)alloc((size_t)NENT * 256 * 2);
    unsigned short* time_bf  = (unsigned short*)alloc((size_t)NTIME * 256 * 2);
    unsigned short* relall_bf= (unsigned short*)alloc((size_t)RELP * 256 * 2);
    unsigned short* hrtab_bf = (unsigned short*)alloc((size_t)RELP * 256 * 2);
    unsigned short* wrts_bf  = (unsigned short*)alloc((size_t)256 * 256 * 2);
    unsigned short* wghh_bf  = (unsigned short*)alloc((size_t)768 * 256 * 2);
    unsigned short* wgih_bf  = (unsigned short*)alloc((size_t)768 * 256 * 2);
    unsigned short* w1_bf    = (unsigned short*)alloc((size_t)256 * 256 * 2);
    unsigned short* w2_bf    = (unsigned short*)alloc((size_t)256 * 256 * 2);
    unsigned short* we_bf    = (unsigned short*)alloc((size_t)256 * 256 * 2);
    unsigned short* wt_bf    = (unsigned short*)alloc((size_t)256 * 256 * 2);
    unsigned short* wh_bf    = (unsigned short*)alloc((size_t)256 * 512 * 2);
    unsigned short* xh_bf    = (unsigned short*)alloc((size_t)Mk * 512 * 2);

    float* hr_tab  = (float*)alloc((size_t)RELP * 256 * 4);
    float* gh_tab  = (float*)alloc((size_t)NPREV * 768 * 4);
    float* gi_tab  = (float*)alloc((size_t)RELP * 768 * 4);
    float* msgA    = (float*)alloc((size_t)NPREV * 256 * 4);
    float* msgB    = (float*)alloc((size_t)RELP * 256 * 4);
    float* htA     = (float*)alloc((size_t)NENT * 256 * 4);
    float* htB     = (float*)alloc((size_t)NTIME * 256 * 4);
    float* qdot    = (float*)alloc(256 * 4);
    float* dmA     = (float*)alloc((size_t)NPREV * 4);
    float* dmB     = (float*)alloc((size_t)RELP * 4);
    float* dhtA    = (float*)alloc((size_t)NENT * 4);
    float* dhtB    = (float*)alloc((size_t)NTIME * 4);
    float* consts  = (float*)alloc(8 * 4);
    float* attention = (float*)alloc((size_t)Ec * 4);
    float* den     = (float*)alloc(256 * 4);
    int*   counts  = (int*)alloc(256 * 4);
    int*   starts  = (int*)alloc(256 * 4);
    int*   cursor  = (int*)alloc(256 * 4);
    int*   bucket  = (int*)alloc((size_t)Ec * 4);
    int*   sel     = (int*)alloc((size_t)Mk * 4);
    float* att_sel = (float*)alloc((size_t)Mk * 4);
    int*   b_sel   = (int*)alloc((size_t)Mk * 4);
    float* att_nrm = (float*)alloc((size_t)Mk * 4);
    float* hrnew   = (float*)alloc((size_t)Mk * 256 * 4);
    float* msg_sel = (float*)alloc((size_t)Mk * 256 * 4);
    float* msg_agg = (float*)alloc((size_t)Mk * 256 * 4);
    unsigned long long* keys = (unsigned long long*)alloc((size_t)Mk * 8);
    int*   tidx    = (int*)alloc((size_t)Mk * 4);
    int*   tailn   = (int*)alloc((size_t)Mk * 3 * 4);

    // d_out layout: tail_nodes(Mk*3) | hidden_node_new(Mk*256) | hidden_rel_out(Mk*256) | att_agg_new(Mk)
    float* out_tail = (float*)d_out;
    float* out_hnn  = out_tail + (size_t)Mk * 3;
    float* out_hro  = out_hnn + (size_t)Mk * 256;
    float* out_agg  = out_hro + (size_t)Mk * 256;

    auto blocks = [](size_t n, int tpb) { return (unsigned)((n + tpb - 1) / tpb); };

    // ---- zero init ----
    zero_f32<<<blocks((size_t)Mk * 516, 256), 256, 0, stream>>>((float*)d_out, (size_t)Mk * 516);
    zero_f32<<<blocks((size_t)Mk * 256, 256), 256, 0, stream>>>(msg_agg, (size_t)Mk * 256);
    zero_f32<<<1, 256, 0, stream>>>(den, 256);
    zero_f32<<<1, 256, 0, stream>>>((float*)counts, 256);
    zero_f32<<<1, 256, 0, stream>>>((float*)cursor, 256);
    zero_f32<<<blocks((size_t)Mk * 3, 256), 256, 0, stream>>>((float*)tailn, (size_t)Mk * 3);

    // ---- bf16 conversions ----
    cvt_bf16<<<blocks((size_t)NPREV * 256, 256), 256, 0, stream>>>(hidden_rel, hrl_bf, (size_t)NPREV * 256);
    cvt_bf16<<<blocks((size_t)NPREV * 256, 256), 256, 0, stream>>>(hidden_node, hnode_bf, (size_t)NPREV * 256);
    cvt_bf16<<<blocks((size_t)NENT * 256, 256), 256, 0, stream>>>(ent_emd, ent_bf, (size_t)NENT * 256);
    cvt_bf16<<<blocks((size_t)NTIME * 256, 256), 256, 0, stream>>>(time_emd, time_bf, (size_t)NTIME * 256);
    build_relall<<<blocks((size_t)RELP * 256, 256), 256, 0, stream>>>(zero_rel, rel_emd, relall_bf);
    cvt_bf16<<<blocks((size_t)256 * 256, 256), 256, 0, stream>>>(lrts_w, wrts_bf, (size_t)256 * 256);
    cvt_bf16<<<blocks((size_t)768 * 256, 256), 256, 0, stream>>>(gwhh, wghh_bf, (size_t)768 * 256);
    cvt_bf16<<<blocks((size_t)768 * 256, 256), 256, 0, stream>>>(gwih, wgih_bf, (size_t)768 * 256);
    cvt_split_bf16<<<blocks((size_t)256 * 256, 256), 256, 0, stream>>>(Wmsg, w1_bf, 256, 256, 512, 0);
    cvt_split_bf16<<<blocks((size_t)256 * 256, 256), 256, 0, stream>>>(Wmsg, w2_bf, 256, 256, 512, 256);
    cvt_split_bf16<<<blocks((size_t)256 * 256, 256), 256, 0, stream>>>(lets_w, we_bf, 256, 256, 512, 0);
    cvt_split_bf16<<<blocks((size_t)256 * 256, 256), 256, 0, stream>>>(lets_w, wt_bf, 256, 256, 512, 256);
    cvt_bf16<<<blocks((size_t)256 * 512, 256), 256, 0, stream>>>(Whw, wh_bf, (size_t)256 * 512);

    auto gemm = [&](const unsigned short* X, const unsigned short* W, const float* bias,
                    float* Y, int M, int N, int K) {
        unsigned mTiles = (unsigned)((M + 15) / 16);
        dim3 grid((mTiles + 7) / 8, (unsigned)(N / 64));
        size_t ldsB = (size_t)64 * (K + BPAD) * 2;   // padded B panel
        gemm_bf16_wmma<<<grid, 256, ldsB, stream>>>(X, W, bias, Y, M, N, K);
    };

    // ---- factored table GEMMs (all WMMA) ----
    gemm(relall_bf, wrts_bf, lrts_b, hr_tab, RELP, 256, 256);        // h_r table
    cvt_bf16<<<blocks((size_t)RELP * 256, 256), 256, 0, stream>>>(hr_tab, hrtab_bf, (size_t)RELP * 256);
    gemm(hrl_bf,   wghh_bf, gbhh, gh_tab, NPREV, 768, 256);          // GRU gh table
    gemm(hrtab_bf, wgih_bf, gbih, gi_tab, RELP, 768, 256);           // GRU gi table
    gemm(hnode_bf, w1_bf, nullptr, msgA, NPREV, 256, 256);           // message part A
    gemm(hrtab_bf, w2_bf, nullptr, msgB, RELP, 256, 256);            // message part B
    gemm(ent_bf,   we_bf, nullptr, htA, NENT, 256, 256);             // h_t ent part
    gemm(time_bf,  wt_bf, nullptr, htB, NTIME, 256, 256);            // h_t time part

    // ---- scalar attention tables ----
    dot_rows<<<blocks(256, 8), 256, 0, stream>>>(query_emd, a1w, qdot, 256);
    dot_rows<<<blocks(NPREV, 8), 256, 0, stream>>>(msgA, a1w + 256, dmA, NPREV);
    dot_rows<<<blocks(RELP, 8), 256, 0, stream>>>(msgB, a1w + 256, dmB, RELP);
    dot_rows<<<blocks(NENT, 8), 256, 0, stream>>>(htA, a1w + 512, dhtA, NENT);
    dot_rows<<<blocks(NTIME, 8), 256, 0, stream>>>(htB, a1w + 512, dhtB, NTIME);
    consts_kernel<<<1, 32, 0, stream>>>(a1w, a1b, Wmsg_b, lets_b, consts);

    // ---- per-edge attention ----
    attention_kernel<<<blocks(Ec, 8), 256, 0, stream>>>(
        facts, gi_tab, gh_tab, hidden_rel, a2w, a2b, qdot, dmA, dmB, dhtA, dhtB,
        consts, attention, Ec);

    // ---- per-batch top-64 ----
    count_kernel<<<blocks(Ec, 256), 256, 0, stream>>>(facts, counts, Ec);
    scan256<<<1, 256, 0, stream>>>(counts, starts);
    scatter_kernel<<<blocks(Ec, 256), 256, 0, stream>>>(facts, starts, cursor, bucket, Ec);
    topk_kernel<<<Bc, 256, 0, stream>>>(bucket, starts, counts, attention, sel);

    // ---- selected-row expansion ----
    selrow_kernel<<<blocks(Mk, 8), 256, 0, stream>>>(
        sel, facts, gi_tab, gh_tab, hidden_rel, msgA, msgB, Wmsg_b, attention, att_agg,
        hrnew, msg_sel, att_sel, b_sel, keys);
    den_kernel<<<blocks(Mk, 256), 256, 0, stream>>>(b_sel, att_sel, den);
    norm_kernel<<<blocks(Mk, 256), 256, 0, stream>>>(b_sel, att_sel, den, att_nrm);

    // ---- sorted unique (single WGP, 196KB LDS) ----
    size_t lds = (size_t)Mk * 8 + (size_t)Mk * 4 + 1024 * 4;
    sort_unique_kernel<<<1, 1024, lds, stream>>>(keys, tidx, tailn, out_tail);

    // ---- segment aggregation ----
    aggregate_kernel<<<blocks(Mk, 8), 256, 0, stream>>>(
        tidx, att_nrm, hrnew, msg_sel, msg_agg, out_hro, out_agg);

    // ---- final projection: hidden_node_new = [msg_agg | tail_emd] @ W_h^T + b ----
    build_xh<<<blocks((size_t)Mk * 256, 256), 256, 0, stream>>>(
        msg_agg, htA, htB, lets_b, tailn, xh_bf);
    gemm(xh_bf, wh_bf, Whb, out_hnn, Mk, 256, 512);
}
